// EquivariantAtomEncoder_32169305047125
// MI455X (gfx1250) — matrix operand order repaired and checked by reading, not compiled, not run on hardware
//
#include <hip/hip_runtime.h>
#include <math.h>

// ---------------------------------------------------------------------------
// EquivariantAtomEncoder for MI455X (gfx1250, wave32, WMMA)
//
//   edge_geom     : sh(E,9) f32, rbf(E,32) f16, fcut(E) f32
//   CSR build     : deg hist -> block scan -> fill (int atomics only)
//   gather_emb    : zf16 = f16(z_emb[z])
//   GEMM x0       : x[:, :128] = (zf16 @ W_in) * mask      (WMMA, f32 acc)
//   per block b (x3):
//     s16  = f16(x[:, :128])
//     proj = s16 @ Wp                   -> f16 (per NODE, not per edge)
//     h    = silu(rbf16 @ rW1 + rb1)    -> f16
//     w    = (h @ rW2 + rb2) * proj[src] * fcut -> f16 (fused epilogue)
//     m    = CSR gather-accumulate of w x sh    -> f16 (no fp atomics)
//     x   += res_scale * (m0@Wo0 | m1@Wo1 | m2@Wo2)
//   irrep_norm    : per-node RMS per irrep block, * mask -> d_out (f32)
//
// Weights are converted f32 -> f16 AND transposed to [N][K] so both WMMA
// fragments come from 128-bit vector loads. GEMM N and K are template
// parameters so all tile loops unroll branch-free with immediate addressing.
// ---------------------------------------------------------------------------

typedef __attribute__((ext_vector_type(16))) _Float16 v16h;
typedef __attribute__((ext_vector_type(8)))  _Float16 v8h;
typedef __attribute__((ext_vector_type(8)))  float    v8f;

#define DIMN 480      // 128 + 3*64 + 5*32

// ---------------------------------------------------------------- utilities
// f32 [nmat][K][N]  ->  f16 transposed [nmat][N][K]
__global__ void conv_bT_k(const float* __restrict__ in, _Float16* __restrict__ out,
                          int K, int N, int nmat) {
  int i = blockIdx.x * blockDim.x + threadIdx.x;
  int total = nmat * K * N;
  if (i >= total) return;
  int m = i / (K * N);
  int r = i - m * (K * N);
  int k = r / N;
  int n = r - k * N;
  out[(long)m * K * N + (long)n * K + k] = (_Float16)in[i];
}

__global__ void gather_emb_k(const int* __restrict__ z, const float* __restrict__ z_emb,
                             _Float16* __restrict__ zf16, int BN) {
  int i = blockIdx.x * blockDim.x + threadIdx.x;
  if (i >= BN * 128) return;
  int n = i >> 7, k = i & 127;
  zf16[i] = (_Float16)z_emb[z[n] * 128 + k];
}

__global__ void extract_scalar_k(const float* __restrict__ x, _Float16* __restrict__ s16, int BN) {
  int i = blockIdx.x * blockDim.x + threadIdx.x;
  if (i >= BN * 128) return;
  int n = i >> 7, k = i & 127;
  s16[i] = (_Float16)x[n * DIMN + k];
}

// ------------------------------------------------------------ edge geometry
__global__ void edge_geom_k(const float* __restrict__ ew, const float* __restrict__ ev,
                            float* __restrict__ sh, float* __restrict__ fcut,
                            _Float16* __restrict__ rbf16, int E) {
  int e = blockIdx.x * blockDim.x + threadIdx.x;
  if (e >= E) return;
  float len = ew[e];
  float inv = 1.0f / fmaxf(len, 1e-8f);
  float X = ev[e * 3 + 0] * inv, Y = ev[e * 3 + 1] * inv, Z = ev[e * 3 + 2] * inv;
  const float s3 = 1.7320508075688772f, s5 = 2.2360679774997896f, s15 = 3.8729833462074170f;
  float* S = sh + (long)e * 9;
  S[0] = 1.0f;
  S[1] = s3 * X;  S[2] = s3 * Y;  S[3] = s3 * Z;
  S[4] = s15 * X * Y;  S[5] = s15 * Y * Z;
  S[6] = 0.5f * s5 * (3.0f * Z * Z - 1.0f);
  S[7] = s15 * X * Z;
  S[8] = 0.5f * s15 * (X * X - Y * Y);
  float dc = fminf(len, 5.0f);
  const float width = 5.0f / 31.0f;
  #pragma unroll
  for (int i = 0; i < 32; ++i) {
    float t = (dc - 5.0f * (float)i / 31.0f) / width;
    rbf16[(long)e * 32 + i] = (_Float16)__expf(-0.5f * t * t);
  }
  fcut[e] = 0.5f * (__cosf(3.14159265358979323f * fminf(len / 5.0f, 1.0f)) + 1.0f);
}

// ---------------------------------------------------------------- CSR build
__global__ void hist_k(const int* __restrict__ dst, int* __restrict__ deg, int E) {
  int e = blockIdx.x * blockDim.x + threadIdx.x;
  if (e < E) atomicAdd(&deg[dst[e]], 1);
}

__global__ void scan_k(const int* __restrict__ deg, int* __restrict__ rowptr,
                       int* __restrict__ cursor, int BN) {
  __shared__ int part[256];
  int t = threadIdx.x;
  int per = (BN + 255) / 256;
  int base = t * per;
  int s = 0;
  for (int i = 0; i < per; ++i) { int idx = base + i; if (idx < BN) s += deg[idx]; }
  part[t] = s;
  __syncthreads();
  for (int off = 1; off < 256; off <<= 1) {
    int v = (t >= off) ? part[t - off] : 0;
    __syncthreads();
    part[t] += v;
    __syncthreads();
  }
  int run = part[t] - s;   // exclusive prefix of this chunk
  for (int i = 0; i < per; ++i) {
    int idx = base + i;
    if (idx < BN) { rowptr[idx] = run; cursor[idx] = run; run += deg[idx]; }
  }
  if (t == 255) rowptr[BN] = run;
}

__global__ void fill_k(const int* __restrict__ dst, int* __restrict__ cursor,
                       int* __restrict__ csr, int E) {
  int e = blockIdx.x * blockDim.x + threadIdx.x;
  if (e >= E) return;
  int p = atomicAdd(&cursor[dst[e]], 1);
  csr[p] = e;
}

// ----------------------------------------------------- generic WMMA GEMM
// C[M x N_] = A[M x K_] * Bt^T (+bias), one wave -> 16 rows x full N_.
// A : f16, element (row,k) at A[row*aRow + k*aK]; AVEC=true requires aK==1
//     and 16-byte-aligned rows (aRow multiple of 8).
// Bt: f16, TRANSPOSED weights [N_][K_]; element (k,n) at Bt[n*K_ + k].
// blockIdx.y = slice index d: A += d*aDStep, out column offset += d*oDStep.
// EPI 0: out f16 = (silu?) val                       -> outH[row*ohRow + col]
// EPI 1: w = val * projn[src[row]*192+col] * fcut    -> outH[row*192 + col]
// EPI 2: outF[row*ofRow + ofOff + col*ofStride] += scale * val
// EPI 3: outF[row*ofRow + col] = val * mask[row]
template <int EPI, bool AVEC, int N_, int K_>
__global__ __launch_bounds__(256) void gemm_wmma_k(
    const _Float16* __restrict__ A, int aRow, int aK, int aDStep,
    const _Float16* __restrict__ Bt,
    int M,
    const float* __restrict__ bias, int actSilu,
    _Float16* __restrict__ outH, int ohRow,
    float* __restrict__ outF, int ofRow, int ofOff, int ofStride, int oDStep,
    const float* __restrict__ scalePtr, int scaleIdx,
    const int* __restrict__ esrc, const float* __restrict__ fcut,
    const _Float16* __restrict__ projn, const float* __restrict__ maskp) {
  constexpr int NT = N_ / 16;   // tiles across N
  constexpr int KC = K_ / 32;   // 32-wide K chunks
  int wave = (blockIdx.x * blockDim.x + threadIdx.x) >> 5;
  int lane = threadIdx.x & 31;
  int rowBase = wave * 16;
  if (rowBase >= M) return;
  int d = blockIdx.y;
  const _Float16* Ad = A + (long)d * aDStep;
  int oOff = ofOff + d * oDStep;

  int lrow = lane & 15;
  int hi = (lane < 16) ? 0 : 1;
  int koA = hi * 8;     // A: hi-half lanes hold K+8 (regs 0-3) / K+24 (regs 4-7)
  int koB = hi * 8;     // B: same K order as A (K+8h regs 0-3, K+16+8h regs 4-7)

  v8f acc[NT];
  #pragma unroll
  for (int t = 0; t < NT; ++t) acc[t] = (v8f){0.f,0.f,0.f,0.f,0.f,0.f,0.f,0.f};

  long arow = (long)(rowBase + lrow) * aRow;
  const _Float16* bcol = Bt + (long)lrow * K_ + koB;   // column lrow, this half's K (table order)

  #pragma unroll
  for (int kc = 0; kc < KC; ++kc) {
    const int kBase = kc * 32;
    v16h a;
    if (AVEC) {
      // two contiguous 8-half chunks: K = kBase+koA..+7 and kBase+16+koA..+23
      v8h lo = *(const v8h*)(Ad + arow + kBase + koA);
      v8h hh = *(const v8h*)(Ad + arow + kBase + 16 + koA);
      a = __builtin_shufflevector(lo, hh, 0,1,2,3,4,5,6,7,8,9,10,11,12,13,14,15);
    } else {
      #pragma unroll
      for (int p = 0; p < 8; ++p) {
        const int kb = (p < 4) ? (koA + 2 * p) : (16 + koA + 2 * (p - 4));
        a[2 * p]     = Ad[arow + (long)(kBase + kb)     * aK];
        a[2 * p + 1] = Ad[arow + (long)(kBase + kb + 1) * aK];
      }
    }
    #pragma unroll
    for (int t = 0; t < NT; ++t) {
      // Bt[t*16 + lrow]: K = kBase+koB..+7 and kBase+16+koB..+23
      const _Float16* bp = bcol + t * 16 * K_ + kBase;
      v8h blo = *(const v8h*)(bp);
      v8h bhi = *(const v8h*)(bp + 16);
      v16h bf = __builtin_shufflevector(blo, bhi,
                  0,1,2,3,4,5,6,7,8,9,10,11,12,13,14,15);
      acc[t] = __builtin_amdgcn_wmma_f32_16x16x32_f16(
          false, a, false, bf, (short)0, acc[t], false, false);
    }
  }

  float scale = scalePtr ? scalePtr[scaleIdx] : 1.0f;
  #pragma unroll
  for (int t = 0; t < NT; ++t) {
    int col = t * 16 + lrow;
    float bv = bias ? bias[col] : 0.0f;
    #pragma unroll
    for (int r = 0; r < 8; ++r) {
      int row = rowBase + r + hi * 8;
      float v = acc[t][r] + bv;
      if (EPI == 0) {
        if (actSilu) v = v / (1.0f + __expf(-v));
        outH[(long)row * ohRow + col] = (_Float16)v;
      } else if (EPI == 1) {
        int s = esrc[row];
        float pv = (float)projn[(long)s * 192 + col];
        v = v * pv * fcut[row];
        outH[(long)row * 192 + col] = (_Float16)v;
      } else if (EPI == 2) {
        long o = (long)row * ofRow + oOff + (long)col * ofStride;
        outF[o] += scale * v;
      } else {
        outF[(long)row * ofRow + col] = v * maskp[row];
      }
    }
  }
}

// ---------------------------------------- CSR gather segment-sum of messages
// 64 threads per node (channel c = tid). m layout per node (576 f16):
//   [0..63]=m0, [64 + c*3 + d]=m1, [256 + c*5 + d]=m2
__global__ __launch_bounds__(64) void gather_messages_k(
    const int* __restrict__ rowptr, const int* __restrict__ csr,
    const _Float16* __restrict__ w16, const float* __restrict__ sh,
    _Float16* __restrict__ m16) {
  int n = blockIdx.x;
  int c = threadIdx.x;
  float a0 = 0.f;
  float a10 = 0.f, a11 = 0.f, a12 = 0.f;
  float a20 = 0.f, a21 = 0.f, a22 = 0.f, a23 = 0.f, a24 = 0.f;
  int s = rowptr[n], e = rowptr[n + 1];
  for (int i = s; i < e; ++i) {
    int ed = csr[i];
    if (i + 1 < e) __builtin_prefetch(&w16[(long)csr[i + 1] * 192], 0, 0);
    const float* S = sh + (long)ed * 9;
    const _Float16* W = w16 + (long)ed * 192;
    float w0 = (float)W[c], w1 = (float)W[64 + c], w2 = (float)W[128 + c];
    a0  += w0 * S[0];
    a10 += w1 * S[1];  a11 += w1 * S[2];  a12 += w1 * S[3];
    a20 += w2 * S[4];  a21 += w2 * S[5];  a22 += w2 * S[6];
    a23 += w2 * S[7];  a24 += w2 * S[8];
  }
  _Float16* M = m16 + (long)n * 576;
  M[c] = (_Float16)a0;
  M[64 + c * 3 + 0] = (_Float16)a10;
  M[64 + c * 3 + 1] = (_Float16)a11;
  M[64 + c * 3 + 2] = (_Float16)a12;
  M[256 + c * 5 + 0] = (_Float16)a20;
  M[256 + c * 5 + 1] = (_Float16)a21;
  M[256 + c * 5 + 2] = (_Float16)a22;
  M[256 + c * 5 + 3] = (_Float16)a23;
  M[256 + c * 5 + 4] = (_Float16)a24;
}

// ------------------------------------------------------ final irrep RMS norm
__global__ void irrep_norm_k(const float* __restrict__ x, const float* __restrict__ mask,
                             float* __restrict__ out, int BN) {
  int wave = (blockIdx.x * blockDim.x + threadIdx.x) >> 5;
  int lane = threadIdx.x & 31;
  if (wave >= BN) return;
  const float* xr = x + (long)wave * DIMN;
  float s0 = 0.f, s1 = 0.f, s2 = 0.f;
  for (int i = lane; i < 128; i += 32) { float v = xr[i];       s0 += v * v; }
  for (int i = lane; i < 192; i += 32) { float v = xr[128 + i]; s1 += v * v; }
  for (int i = lane; i < 160; i += 32) { float v = xr[320 + i]; s2 += v * v; }
  #pragma unroll
  for (int o = 16; o > 0; o >>= 1) {
    s0 += __shfl_xor(s0, o, 32);
    s1 += __shfl_xor(s1, o, 32);
    s2 += __shfl_xor(s2, o, 32);
  }
  float r0 = sqrtf(s0 / 128.0f + 1e-6f);
  float r1 = sqrtf(s1 / 64.0f  + 1e-6f);
  float r2 = sqrtf(s2 / 32.0f  + 1e-6f);
  float mk = mask[wave];
  float* o_ = out + (long)wave * DIMN;
  for (int i = lane; i < 128; i += 32) o_[i]       = xr[i]       / r0 * mk;
  for (int i = lane; i < 192; i += 32) o_[128 + i] = xr[128 + i] / r1 * mk;
  for (int i = lane; i < 160; i += 32) o_[320 + i] = xr[320 + i] / r2 * mk;
}

// ---------------------------------------------------------------------------
static inline size_t alignup(size_t x) { return (x + 255) & ~(size_t)255; }

extern "C" void kernel_launch(void* const* d_in, const int* in_sizes, int n_in,
                              void* d_out, int out_size, void* d_ws, size_t ws_size,
                              hipStream_t stream) {
  const int BN = in_sizes[0];   // 4096
  const int E  = in_sizes[2];   // 131072

  const int*   z        = (const int*)  d_in[0];
  const float* mask     = (const float*)d_in[1];
  const int*   esrc     = (const int*)  d_in[2];
  const int*   edst     = (const int*)  d_in[3];
  const float* ew       = (const float*)d_in[4];
  const float* ev       = (const float*)d_in[5];
  const float* z_emb    = (const float*)d_in[6];
  const float* W_in     = (const float*)d_in[7];
  const float* Wp       = (const float*)d_in[8];
  const float* rW1      = (const float*)d_in[9];
  const float* rb1      = (const float*)d_in[10];
  const float* rW2      = (const float*)d_in[11];
  const float* rb2      = (const float*)d_in[12];
  const float* Wo0      = (const float*)d_in[13];
  const float* Wo1      = (const float*)d_in[14];
  const float* Wo2      = (const float*)d_in[15];
  const float* resScale = (const float*)d_in[16];
  float* out = (float*)d_out;

  // -------- workspace carve-out
  char* w = (char*)d_ws;
  auto alloc = [&](size_t bytes) { char* p = w; w += alignup(bytes); return p; };
  float*    x       = (float*)   alloc((size_t)BN * DIMN * 4);
  float*    sh      = (float*)   alloc((size_t)E * 9 * 4);
  float*    fcut    = (float*)   alloc((size_t)E * 4);
  _Float16* rbf16   = (_Float16*)alloc((size_t)E * 32 * 2);
  _Float16* h16     = (_Float16*)alloc((size_t)E * 64 * 2);
  _Float16* w16     = (_Float16*)alloc((size_t)E * 192 * 2);
  _Float16* zf16    = (_Float16*)alloc((size_t)BN * 128 * 2);
  _Float16* s16     = (_Float16*)alloc((size_t)BN * 128 * 2);
  _Float16* projn16 = (_Float16*)alloc((size_t)BN * 192 * 2);
  _Float16* m16     = (_Float16*)alloc((size_t)BN * 576 * 2);
  _Float16* Win16   = (_Float16*)alloc((size_t)128 * 128 * 2);       // [N][K]
  _Float16* Wp16    = (_Float16*)alloc((size_t)3 * 128 * 192 * 2);   // [192][128]
  _Float16* rW116   = (_Float16*)alloc((size_t)3 * 32 * 64 * 2);     // [64][32]
  _Float16* rW216   = (_Float16*)alloc((size_t)3 * 64 * 192 * 2);    // [192][64]
  _Float16* Wo016   = (_Float16*)alloc((size_t)3 * 64 * 128 * 2);    // [128][64]
  _Float16* Wo116   = (_Float16*)alloc((size_t)3 * 64 * 64 * 2);     // [64][64]
  _Float16* Wo216   = (_Float16*)alloc((size_t)3 * 64 * 32 * 2);     // [32][64]
  int*      deg     = (int*)     alloc((size_t)BN * 4);
  int*      cursor  = (int*)     alloc((size_t)BN * 4);
  int*      rowptr  = (int*)     alloc((size_t)(BN + 1) * 4);
  int*      csr     = (int*)     alloc((size_t)E * 4);
  (void)ws_size; (void)n_in; (void)out_size;

  auto g1 = [](int n) { return dim3((n + 255) / 256); };

  // -------- weight conversion f32 -> f16, transposed to [N][K]
  conv_bT_k<<<g1(128 * 128), 256, 0, stream>>>(W_in, Win16, 128, 128, 1);
  conv_bT_k<<<g1(3 * 128 * 192), 256, 0, stream>>>(Wp, Wp16, 128, 192, 3);
  conv_bT_k<<<g1(3 * 32 * 64), 256, 0, stream>>>(rW1, rW116, 32, 64, 3);
  conv_bT_k<<<g1(3 * 64 * 192), 256, 0, stream>>>(rW2, rW216, 64, 192, 3);
  conv_bT_k<<<g1(3 * 64 * 128), 256, 0, stream>>>(Wo0, Wo016, 64, 128, 3);
  conv_bT_k<<<g1(3 * 64 * 64), 256, 0, stream>>>(Wo1, Wo116, 64, 64, 3);
  conv_bT_k<<<g1(3 * 64 * 32), 256, 0, stream>>>(Wo2, Wo216, 64, 32, 3);

  // -------- edge geometry + CSR of edge_dst
  edge_geom_k<<<g1(E), 256, 0, stream>>>(ew, ev, sh, fcut, rbf16, E);
  hipMemsetAsync(deg, 0, (size_t)BN * 4, stream);
  hist_k<<<g1(E), 256, 0, stream>>>(edst, deg, E);
  scan_k<<<1, 256, 0, stream>>>(deg, rowptr, cursor, BN);
  fill_k<<<g1(E), 256, 0, stream>>>(edst, cursor, csr, E);

  // -------- x init: x[:, :128] = (z_emb[z] @ W_in) * mask, rest zero
  gather_emb_k<<<g1(BN * 128), 256, 0, stream>>>(z, z_emb, zf16, BN);
  hipMemsetAsync(x, 0, (size_t)BN * DIMN * 4, stream);

  auto gemmGrid = [](int M) { return dim3((M / 16 + 7) / 8); };
  auto gemmGridY = [](int M, int y) { return dim3((M / 16 + 7) / 8, y); };

  gemm_wmma_k<3, true, 128, 128><<<gemmGrid(BN), 256, 0, stream>>>(
      zf16, 128, 1, 0, Win16, BN,
      nullptr, 0, nullptr, 0, x, DIMN, 0, 1, 0,
      nullptr, 0, nullptr, nullptr, nullptr, mask);

  // -------- message-passing blocks
  for (int b = 0; b < 3; ++b) {
    extract_scalar_k<<<g1(BN * 128), 256, 0, stream>>>(x, s16, BN);

    // per-node projection: proj = s @ Wp[b]  (BN x 192)
    gemm_wmma_k<0, true, 192, 128><<<gemmGrid(BN), 256, 0, stream>>>(
        s16, 128, 1, 0, Wp16 + (size_t)b * 128 * 192, BN,
        nullptr, 0, projn16, 192, nullptr, 0, 0, 0, 0,
        nullptr, 0, nullptr, nullptr, nullptr, nullptr);

    // radial MLP layer 1: h = silu(rbf @ rW1 + rb1)  (E x 64)
    gemm_wmma_k<0, true, 64, 32><<<gemmGrid(E), 256, 0, stream>>>(
        rbf16, 32, 1, 0, rW116 + (size_t)b * 32 * 64, E,
        rb1 + (size_t)b * 64, 1, h16, 64, nullptr, 0, 0, 0, 0,
        nullptr, 0, nullptr, nullptr, nullptr, nullptr);

    // radial MLP layer 2 fused with w = r * proj[src] * fcut  (E x 192)
    gemm_wmma_k<1, true, 192, 64><<<gemmGrid(E), 256, 0, stream>>>(
        h16, 64, 1, 0, rW216 + (size_t)b * 64 * 192, E,
        rb2 + (size_t)b * 192, 0, w16, 192, nullptr, 0, 0, 0, 0,
        nullptr, 0, esrc, fcut, projn16, nullptr);

    // segment-sum via CSR gather: m (BN x 576)
    gather_messages_k<<<dim3(BN), 64, 0, stream>>>(rowptr, csr, w16, sh, m16);

    // x[:, 0:128] += rs * m0 @ Wo0   (contiguous A -> vector loads)
    gemm_wmma_k<2, true, 128, 64><<<gemmGrid(BN), 256, 0, stream>>>(
        m16, 576, 1, 0, Wo016 + (size_t)b * 64 * 128, BN,
        nullptr, 0, nullptr, 0, x, DIMN, 0, 1, 0,
        resScale, b, nullptr, nullptr, nullptr, nullptr);
    // x[:, 128 + k*3 + d] += rs * m1[:, :, d] @ Wo1   (strided A, d = blockIdx.y)
    gemm_wmma_k<2, false, 64, 64><<<gemmGridY(BN, 3), 256, 0, stream>>>(
        m16 + 64, 576, 3, 1, Wo116 + (size_t)b * 64 * 64, BN,
        nullptr, 0, nullptr, 0, x, DIMN, 128, 3, 1,
        resScale, b, nullptr, nullptr, nullptr, nullptr);
    // x[:, 320 + k*5 + d] += rs * m2[:, :, d] @ Wo2   (strided A, d = blockIdx.y)
    gemm_wmma_k<2, false, 32, 64><<<gemmGridY(BN, 5), 256, 0, stream>>>(
        m16 + 256, 576, 5, 1, Wo216 + (size_t)b * 64 * 32, BN,
        nullptr, 0, nullptr, 0, x, DIMN, 320, 5, 1,
        resScale, b, nullptr, nullptr, nullptr, nullptr);
  }

  // -------- final per-irrep RMS norm * mask -> out (f32)
  irrep_norm_k<<<dim3((BN + 7) / 8), 256, 0, stream>>>(x, mask, out, BN);
}
